// Tanh_RNN_25116968747218
// MI455X (gfx1250) — matrix operand order repaired and checked by reading, not compiled, not run on hardware
//
#include <hip/hip_runtime.h>
#include <hip/hip_bf16.h>
#include <stdint.h>

// ---------------------------------------------------------------------------
// Tanh RNN for MI455X (gfx1250).
//  Phase 1: xw[t,b,u] = (emb[sentence] @ Wx) + b  as one big bf16 WMMA GEMM.
//  Phase 2: 512-step recurrent scan, Wh resident in LDS (bf16, fragment-
//           swizzled, split over 8 WGPs), h exchanged via L2 ping-pong +
//           atomic grid barrier, async-to-LDS copies overlap WMMA compute.
//  Phase 3: tiny MLP head, single block.
// ---------------------------------------------------------------------------

typedef __attribute__((ext_vector_type(16))) __bf16 v16bf;
typedef __attribute__((ext_vector_type(8)))  __bf16 v8bf;
typedef __attribute__((ext_vector_type(8)))  float  v8f;

#define T_LEN 512
#define BATCH 64
#define EMB   256
#define UNITS 512
#define NWG   8
#define COLS_PER_WG 64   // UNITS / NWG

#if __has_builtin(__builtin_amdgcn_tanhf)
#define FAST_TANH(x) __builtin_amdgcn_tanhf(x)
#else
#define FAST_TANH(x) tanhf(x)
#endif

// ------------------------- CDNA5 helpers -----------------------------------

static __device__ __forceinline__ v8f wmma_bf16f32(v16bf a, v16bf b, v8f c) {
  // (neg_a, A, neg_b, B, c_mod, C, reuse_a, reuse_b)
  return __builtin_amdgcn_wmma_f32_16x16x32_bf16(false, a, false, b, (short)0,
                                                 c, false, false);
}

// Async copy global -> LDS (128b), tracked with ASYNCcnt (CDNA5 ISA).
static __device__ __forceinline__ void async_load_b128(uint32_t lds_addr,
                                                       uint64_t gbase,
                                                       uint32_t voff) {
  asm volatile("global_load_async_to_lds_b128 %0, %1, %2 offset:0"
               :: "v"(lds_addr), "v"(voff), "s"(gbase)
               : "memory");
}
static __device__ __forceinline__ void wait_async0() {
  asm volatile("s_wait_asynccnt 0" ::: "memory");
}

// WMMA 16-bit fragment maps (CDNA5 ISA §7.12.2), wave32:
//  A (16x32): row = lane&15; element i -> k = (i<8 ? hi*8+i : 16+hi*8+i-8)
//             i.e. two contiguous 8-element (16B) chunks.
//  B (32x16): col = lane&15; k map as A; this code loads both A and B with k = hi*16 + i.
//  C/D      : VGPR v -> row v + hi*8, col = lane&15.      (hi = lane>>4)

// Build v16bf from two contiguous 8-element LDS chunks.
static __device__ __forceinline__ v16bf load_a_frag(const __bf16* base_lo,
                                                    const __bf16* base_hi) {
  v8bf lo = *(const v8bf*)base_lo;
  v8bf hi = *(const v8bf*)base_hi;
  return __builtin_shufflevector(lo, hi, 0, 1, 2, 3, 4, 5, 6, 7, 8, 9, 10, 11,
                                 12, 13, 14, 15);
}

// ------------------------- Phase 0: prep -----------------------------------
__global__ __launch_bounds__(256) void prep_kernel(const float* __restrict__ Wh,
                                                   __bf16* __restrict__ whb,
                                                   int* __restrict__ ctr) {
  int i = blockIdx.x * 256 + threadIdx.x;
  if (i < UNITS * UNITS) whb[i] = (__bf16)Wh[i];
  if (i < T_LEN) ctr[i] = 0;
}

// ------------------------- Phase 1: xw GEMM --------------------------------
// One wave per 16x16 output tile. M = T*B = 32768 (row r = t*64 + b),
// K = 256, N = 512. Gathers embedding rows, converts to bf16 in-register.
__global__ __launch_bounds__(256) void xw_gemm_kernel(
    const int* __restrict__ sent, const float* __restrict__ emb,
    const float* __restrict__ Wx, const float* __restrict__ bias,
    __bf16* __restrict__ xw) {
  int wid    = blockIdx.x * 8 + (threadIdx.x >> 5);
  int lane   = threadIdx.x & 31;
  int hi     = (lane >> 4) & 1;
  int tile_m = wid >> 5;                 // 32 N-tiles
  int tile_n = wid & 31;
  int mrow   = tile_m * 16 + (lane & 15);
  int b      = mrow & (BATCH - 1);
  int t      = mrow >> 6;
  int tok    = sent[b * T_LEN + t];
  const float* arow = emb + (size_t)tok * EMB;
  int ncol   = tile_n * 16 + (lane & 15);

  v8f acc = {};
  for (int kb = 0; kb < EMB; kb += 32) {
    // A: two contiguous 8-float chunks -> b128 loads.
    const float4* a0 = (const float4*)&arow[kb + hi * 16];
    const float4* a1 = (const float4*)&arow[kb + hi * 16 + 8];
    float4 x0 = a0[0], x1 = a0[1], x2 = a1[0], x3 = a1[1];
    v16bf af;
    af[0] = (__bf16)x0.x;  af[1] = (__bf16)x0.y;
    af[2] = (__bf16)x0.z;  af[3] = (__bf16)x0.w;
    af[4] = (__bf16)x1.x;  af[5] = (__bf16)x1.y;
    af[6] = (__bf16)x1.z;  af[7] = (__bf16)x1.w;
    af[8] = (__bf16)x2.x;  af[9] = (__bf16)x2.y;
    af[10] = (__bf16)x2.z; af[11] = (__bf16)x2.w;
    af[12] = (__bf16)x3.x; af[13] = (__bf16)x3.y;
    af[14] = (__bf16)x3.z; af[15] = (__bf16)x3.w;
    v16bf bf;
#pragma unroll
    for (int i = 0; i < 16; ++i) {   // B: k = hi*16 + i
      bf[i] = (__bf16)Wx[(size_t)(kb + hi * 16 + i) * UNITS + ncol];
    }
    acc = wmma_bf16f32(af, bf, acc);
  }
  float bv = bias[ncol];
#pragma unroll
  for (int v = 0; v < 8; ++v) {
    int r = tile_m * 16 + v + hi * 8;
    xw[(size_t)r * UNITS + ncol] = (__bf16)(acc[v] + bv);
  }
}

// ------------------------- Phase 2: recurrent scan -------------------------
// 8 persistent workgroups x 512 threads (16 waves). WG g owns output columns
// [g*64, g*64+64). Wh slice lives in LDS, pre-swizzled into B-fragment order
// [kblock(16)][ntile(4)][lane(32)][elem(16)] so each lane's fragment is one
// contiguous 32B read.
__global__ __launch_bounds__(512) void rnn_scan_kernel(
    const __bf16* __restrict__ xw, const __bf16* __restrict__ whb,
    __bf16* __restrict__ hbuf,     // [2][64][512] ping-pong
    float* __restrict__ hfinal,    // [64][512]
    int* __restrict__ ctr) {
  __shared__ __align__(32) __bf16 lds_whf[UNITS / 32][4][32][16];   // 64 KB
  __shared__ __align__(32) __bf16 lds_h[BATCH * UNITS];             // 64 KB
  __shared__ __align__(32) __bf16 lds_xw[2][BATCH * COLS_PER_WG];   // 16 KB

  const int g    = blockIdx.x;
  const int tid  = threadIdx.x;
  const int lane = tid & 31;
  const int hi   = (lane >> 4) & 1;
  const int wv   = tid >> 5;       // 16 waves
  const int tm   = wv >> 2;        // 4x4 tiles of 16x16 over the 64x64 slab
  const int tn   = wv & 3;
  const uint64_t xw_base = (uint64_t)(uintptr_t)xw;

  // Fill swizzled Wh slice (once): element (kbb, tnn, ln, i) holds
  // Wh[kbb*32 + (ln>>4)*16 + i][g*64 + tnn*16 + (ln&15)].
  for (int idx = tid; idx < (UNITS / 32) * 4 * 32 * 16; idx += 512) {
    int i   = idx & 15;
    int ln  = (idx >> 4) & 31;
    int tnn = (idx >> 9) & 3;
    int kbb = idx >> 11;
    int k   = kbb * 32 + ((ln >> 4) & 1) * 16 + i;
    int n   = g * COLS_PER_WG + tnn * 16 + (ln & 15);
    lds_whf[kbb][tnn][ln][i] = whb[(size_t)k * UNITS + n];
  }
  // h0 = 0.
  for (int idx = tid; idx < BATCH * UNITS; idx += 512)
    lds_h[idx] = (__bf16)0.0f;
  // Async prefetch xw tile for t = 0: 512 threads x 16B.
  {
    int m = tid >> 3, seg = tid & 7;
    uint32_t ldsa = (uint32_t)(uintptr_t)&lds_xw[0][m * COLS_PER_WG + seg * 8];
    uint32_t voff =
        (uint32_t)((m * UNITS + g * COLS_PER_WG + seg * 8) * 2);
    async_load_b128(ldsa, xw_base, voff);
  }
  wait_async0();
  __syncthreads();

  const __bf16* bfrag_base = &lds_whf[0][tn][lane][0];
  const int arow = tm * 16 + (lane & 15);
  const int nl   = tn * 16 + (lane & 15);

  for (int t = 0; t < T_LEN; ++t) {
    const int buf = t & 1;

    // Prefetch next xw tile while we compute this step.
    if (t + 1 < T_LEN) {
      int m = tid >> 3, seg = tid & 7;
      uint32_t ldsa =
          (uint32_t)(uintptr_t)&lds_xw[buf ^ 1][m * COLS_PER_WG + seg * 8];
      uint32_t voff = (uint32_t)((((uint32_t)(t + 1) * BATCH + m) * UNITS +
                                  g * COLS_PER_WG + seg * 8) * 2);
      async_load_b128(ldsa, xw_base, voff);
    }

    // acc starts at xw_t (bias folded in already); add h @ Wh_slice.
    v8f acc;
#pragma unroll
    for (int v = 0; v < 8; ++v)
      acc[v] = (float)lds_xw[buf][(tm * 16 + v + hi * 8) * COLS_PER_WG + nl];

#pragma unroll 4
    for (int kb = 0; kb < UNITS; kb += 32) {
      v16bf af = load_a_frag(&lds_h[arow * UNITS + kb + hi * 16],
                             &lds_h[arow * UNITS + kb + hi * 16 + 8]);
      v16bf bf = *(const v16bf*)(bfrag_base + (size_t)(kb >> 5) * (4 * 32 * 16));
      acc = wmma_bf16f32(af, bf, acc);
    }

    // tanh, publish our 64x64 slice to the ping-pong buffer.
    __bf16* hout = hbuf + (size_t)buf * BATCH * UNITS;
    const int gcol = g * COLS_PER_WG + nl;
#pragma unroll
    for (int v = 0; v < 8; ++v) {
      float hv = FAST_TANH(acc[v]);
      int r = tm * 16 + v + hi * 8;
      hout[(size_t)r * UNITS + gcol] = (__bf16)hv;
      if (t == T_LEN - 1) hfinal[(size_t)r * UNITS + gcol] = hv;
    }

    // Grid barrier (one counter per step; agent scope).
    __threadfence();
    __syncthreads();
    if (tid == 0) {
      __hip_atomic_fetch_add(&ctr[t], 1, __ATOMIC_ACQ_REL,
                             __HIP_MEMORY_SCOPE_AGENT);
      while (__hip_atomic_load(&ctr[t], __ATOMIC_ACQUIRE,
                               __HIP_MEMORY_SCOPE_AGENT) < NWG) {
      }
    }
    __syncthreads();

    // Async reload of full h (64 KB) into LDS: 512 threads x 128B.
    {
      int row = tid >> 3, seg = tid & 7;
      uint32_t base_elem = (uint32_t)(row * UNITS + seg * 64);
      uint64_t hsrc = (uint64_t)(uintptr_t)(hbuf + (size_t)buf * BATCH * UNITS);
#pragma unroll
      for (int c = 0; c < 8; ++c) {
        uint32_t ldsa = (uint32_t)(uintptr_t)&lds_h[base_elem + c * 8];
        async_load_b128(ldsa, hsrc, (base_elem + c * 8) * 2);
      }
    }
    wait_async0();
    __syncthreads();
  }
}

// ------------------------- Phase 3: MLP head -------------------------------
__global__ __launch_bounds__(256) void head_kernel(
    const float* __restrict__ hfinal, const float* __restrict__ W1,
    const float* __restrict__ b1, const float* __restrict__ W2,
    const float* __restrict__ b2, const float* __restrict__ W3,
    const float* __restrict__ b3, float* __restrict__ out) {
  __shared__ float z1[BATCH * 128];
  __shared__ float z2[BATCH * 64];
  int tid = threadIdx.x;
  for (int o = tid; o < BATCH * 128; o += 256) {
    int bb = o >> 7, j = o & 127;
    float s = b1[j];
    for (int k = 0; k < UNITS; ++k) s += hfinal[bb * UNITS + k] * W1[k * 128 + j];
    z1[o] = fmaxf(s, 0.0f);
  }
  __syncthreads();
  for (int o = tid; o < BATCH * 64; o += 256) {
    int bb = o >> 6, j = o & 63;
    float s = b2[j];
    for (int k = 0; k < 128; ++k) s += z1[bb * 128 + k] * W2[k * 64 + j];
    z2[o] = fmaxf(s, 0.0f);
  }
  __syncthreads();
  if (tid < BATCH) {
    float s = b3[0];
    for (int k = 0; k < 64; ++k) s += z2[tid * 64 + k] * W3[k];
    out[tid] = 1.0f / (1.0f + expf(-s));
  }
}

// ------------------------- launcher ----------------------------------------
extern "C" void kernel_launch(void* const* d_in, const int* in_sizes, int n_in,
                              void* d_out, int out_size, void* d_ws,
                              size_t ws_size, hipStream_t stream) {
  const int*   sent = (const int*)d_in[0];
  const float* emb  = (const float*)d_in[1];
  const float* Wx   = (const float*)d_in[2];
  const float* Wh   = (const float*)d_in[3];
  const float* bias = (const float*)d_in[4];
  const float* W1   = (const float*)d_in[5];
  const float* b1   = (const float*)d_in[6];
  const float* W2   = (const float*)d_in[7];
  const float* b2   = (const float*)d_in[8];
  const float* W3   = (const float*)d_in[9];
  const float* b3   = (const float*)d_in[10];

  char* ws = (char*)d_ws;
  __bf16* xw   = (__bf16*)(ws);                 // 32768*512*2  = 33,554,432 B
  __bf16* whb  = (__bf16*)(ws + 33554432);      // 512*512*2    =    524,288 B
  __bf16* hbuf = (__bf16*)(ws + 34078720);      // 2*64*512*2   =    131,072 B
  float*  hfin = (float*)(ws + 34209792);       // 64*512*4     =    131,072 B
  int*    ctr  = (int*)(ws + 34340864);         // 512*4        =      2,048 B

  prep_kernel<<<1024, 256, 0, stream>>>(Wh, whb, ctr);
  xw_gemm_kernel<<<8192, 256, 0, stream>>>(sent, emb, Wx, bias, xw);
  rnn_scan_kernel<<<NWG, 512, 0, stream>>>(xw, whb, hbuf, hfin, ctr);
  head_kernel<<<1, 256, 0, stream>>>(hfin, W1, b1, W2, b2, W3, b3,
                                     (float*)d_out);
}
